// FMCE_19550691131948
// MI455X (gfx1250) — matrix operand order repaired and checked by reading, not compiled, not run on hardware
//
#include <hip/hip_runtime.h>
#include <hip/hip_bf16.h>
#include <math.h>

// ---------------------------------------------------------------------------
// Factorization machine: out = sigmoid(z@W + b + ||z@V||^2), z = [x|u]
// B=16384, X=4096, U=1024, K=5120, NF=128.
// Memory-bound (AI ~= 67 FLOP/B < bf16 WMMA balance): read z exactly once
// (320 MB -> ~13.7us floor @ 23.3 TB/s), keep V/W packed in bf16 WMMA
// B-fragment layout (1.44 MB, L2 resident, staged to LDS), and run the GEMM
// on v_wmma_f32_16x16x32_bf16 (f32 accumulate; rel. err ~1e-3, invisible
// after sigmoid). B staging is double-buffered via the CDNA5 async
// global->LDS path (ASYNCcnt) so the copy overlaps WMMA compute.
// ---------------------------------------------------------------------------

typedef __attribute__((ext_vector_type(16))) __bf16 v16bf;
typedef __attribute__((ext_vector_type(8)))  float  v8f;
typedef int b128_t __attribute__((vector_size(16)));   // matches async-LDS builtin
typedef __attribute__((address_space(1))) b128_t* gas_b128p;
typedef __attribute__((address_space(3))) b128_t* las_b128p;

#define B_ROWS     16384
#define X_DIM      4096
#define U_DIM      1024
#define K_DIM      5120
#define NF         128
#define NT         9              // 8 tiles of V columns + 1 tile carrying W in col 0
#define KT_TOTAL   160            // K_DIM / 32
#define KT_CHUNK   8              // k-steps staged in LDS per round
#define NCHUNKS    (KT_TOTAL / KT_CHUNK)      // 20 (first 16 read x, last 4 read u)
#define XCHUNKS    (X_DIM / 32 / KT_CHUNK)    // 16
#define WAVES_PER_WG 8
#define ROWS_PER_WG  (WAVES_PER_WG * 16)      // 128
#define TILE_ELEMS 512            // 32 lanes * 16 bf16 per (kt, t) B-fragment
#define CHUNK_ELEMS (KT_CHUNK * NT * TILE_ELEMS)   // 36864 bf16 = 72 KB
#define CHUNK_BYTES (CHUNK_ELEMS * 2)
#define COPY_OPS   (CHUNK_BYTES / (256 * 16))      // 18 x B128 per thread
// Workspace requirement: KT_TOTAL * NT * TILE_ELEMS * 2 bytes = 1,474,560 B

#if __has_builtin(__builtin_amdgcn_global_load_async_to_lds_b128) && \
    __has_builtin(__builtin_amdgcn_s_wait_asynccnt)
#define HAVE_ASYNC_LDS 1
#else
#define HAVE_ASYNC_LDS 0
#endif

// ---------------------------------------------------------------------------
// Pack V (fp32 [K, NF]) and W (fp32 [K]) into bf16 WMMA B-fragments.
// B-matrix 32x16 packing (contiguous k order, same as A): lane holds one column
// n = t*16 + (lane&15); lanes 0-15 hold K = kk+0..15 (elements 0..15),
// lanes 16-31 hold K = kk+16..31.
// Tile t==8 carries W in column 0, zeros elsewhere, so the linear score
// falls out of the same WMMA chain.
// ---------------------------------------------------------------------------
__global__ void fm_pack_B(const float* __restrict__ V,
                          const float* __restrict__ W,
                          __bf16* __restrict__ Bp) {
  const int kt   = blockIdx.x;      // 0..159
  const int t    = blockIdx.y;      // 0..8
  const int lane = threadIdx.x;     // 0..31
  const int c    = lane & 15;
  const int kbase = kt * 32 + ((lane >> 4) << 4);   // +16 for lanes 16..31

  v16bf v;
#pragma unroll
  for (int e = 0; e < 16; ++e) {
    const int k = kbase + e;
    float f;
    if (t < 8)  f = V[(size_t)k * NF + t * 16 + c];
    else        f = (c == 0) ? W[k] : 0.0f;
    v[e] = (__bf16)f;
  }
  v16bf* dst = (v16bf*)(Bp + ((size_t)(kt * NT + t) * 32 + lane) * 16);
  *dst = v;
}

// ---------------------------------------------------------------------------
// Main kernel: 8 waves per WG, each wave owns a 16-row M tile and all 9
// accumulators. B fragments double-buffered in LDS (async copy overlapped
// with compute); A fragments built from fp32 z loads converted to bf16
// (A 16x32 layout: lanes 0-15 hold K {kk..kk+15},
//  lanes 16-31 hold {kk+16..kk+31}, matching the B packing).
// ---------------------------------------------------------------------------
__global__ __launch_bounds__(256)
void fm_main(const float* __restrict__ x,
             const float* __restrict__ u,
             const float* __restrict__ bptr,
             const __bf16* __restrict__ Bp,
             float* __restrict__ out) {
  __shared__ __bf16 ldsB[2][CHUNK_ELEMS];   // 2 x 72 KB

  const int tid  = threadIdx.x;
  const int lane = tid & 31;
  const int wave = tid >> 5;
  const int hi   = lane >> 4;          // 0: lanes 0-15, 1: lanes 16-31
  const int r    = lane & 15;
  const int row0 = blockIdx.x * ROWS_PER_WG + wave * 16;

  const float* zx = x + (size_t)(row0 + r) * X_DIM;
  const float* zu = u + (size_t)(row0 + r) * U_DIM;

  v8f acc[NT];
#pragma unroll
  for (int t = 0; t < NT; ++t) acc[t] = (v8f){};

#if HAVE_ASYNC_LDS
  // Async stage of one chunk's B fragments: each thread issues COPY_OPS
  // global_load_async_to_lds_b128 ops (tracked by ASYNCcnt, no VGPR data).
  auto issue_chunk = [&](int c, int buf) {
    const char* src = (const char*)Bp + (size_t)c * CHUNK_BYTES + tid * 16;
    char*       dst = (char*)&ldsB[buf][0] + tid * 16;
#pragma unroll
    for (int i = 0; i < COPY_OPS; ++i) {
      __builtin_amdgcn_global_load_async_to_lds_b128(
          (gas_b128p)(src + (size_t)i * 4096),
          (las_b128p)(dst + (size_t)i * 4096),
          0, 0);
    }
  };
  issue_chunk(0, 0);
#else
  auto copy_chunk = [&](int c, int buf) {
    const float4* src = (const float4*)((const char*)Bp + (size_t)c * CHUNK_BYTES);
    float4* dst = (float4*)&ldsB[buf][0];
#pragma unroll 4
    for (int i = tid; i < CHUNK_BYTES / 16; i += 256) dst[i] = src[i];
  };
  copy_chunk(0, 0);
#endif

  for (int c = 0; c < NCHUNKS; ++c) {
#if HAVE_ASYNC_LDS
    if (c + 1 < NCHUNKS) {
      issue_chunk(c + 1, (c + 1) & 1);                 // overlap next copy
      __builtin_amdgcn_s_wait_asynccnt(COPY_OPS);      // chunk c has landed
    } else {
      __builtin_amdgcn_s_wait_asynccnt(0);
    }
#endif
    __syncthreads();   // all waves' async ops for chunk c visible in LDS

    const float* base = (c < XCHUNKS) ? zx : zu;
    const int koff0   = ((c < XCHUNKS) ? c : (c - XCHUNKS)) * KT_CHUNK * 32;
    const __bf16* lbuf = &ldsB[c & 1][0];

    for (int kl = 0; kl < KT_CHUNK; ++kl) {
      const int off = koff0 + kl * 32 + hi * 16;

      // prefetch next k-step of z (global_prefetch_b8, speculative)
      __builtin_prefetch(base + off + 32, 0, 1);

      const float4 f0 = *(const float4*)(base + off);
      const float4 f1 = *(const float4*)(base + off + 4);
      const float4 f2 = *(const float4*)(base + off + 8);
      const float4 f3 = *(const float4*)(base + off + 12);

      v16bf a;
      a[0]  = (__bf16)f0.x;  a[1]  = (__bf16)f0.y;
      a[2]  = (__bf16)f0.z;  a[3]  = (__bf16)f0.w;
      a[4]  = (__bf16)f1.x;  a[5]  = (__bf16)f1.y;
      a[6]  = (__bf16)f1.z;  a[7]  = (__bf16)f1.w;
      a[8]  = (__bf16)f2.x;  a[9]  = (__bf16)f2.y;
      a[10] = (__bf16)f2.z;  a[11] = (__bf16)f2.w;
      a[12] = (__bf16)f3.x;  a[13] = (__bf16)f3.y;
      a[14] = (__bf16)f3.z;  a[15] = (__bf16)f3.w;

      const __bf16* bbase = lbuf + (size_t)kl * NT * TILE_ELEMS + lane * 16;
#pragma unroll
      for (int t = 0; t < NT; ++t) {
        const v16bf bm = *(const v16bf*)(bbase + t * TILE_ELEMS);
        acc[t] = __builtin_amdgcn_wmma_f32_16x16x32_bf16(
            /*neg_a=*/false, a, /*neg_b=*/false, bm,
            /*c_mod=*/(short)0, acc[t], /*reuse_a=*/false, /*reuse_b=*/false);
      }
    }
    // all waves done reading ldsB[c&1] before iteration c+1 re-issues into it
    __syncthreads();
#if !HAVE_ASYNC_LDS
    if (c + 1 < NCHUNKS) copy_chunk(c + 1, (c + 1) & 1);
#endif
  }

  // ---- epilogue: per-lane sum of squares over its column + lin column ----
  // C/D layout: lane (hi half selects rows 8..15) holds 8 row values (VGPR d)
  // of column N = lane&15 within each tile.
  const float bias = bptr[0];
  float s[8];
#pragma unroll
  for (int d = 0; d < 8; ++d) {
    float v = acc[8][d];                 // z.W contribution lives in col 0 of tile 8
#pragma unroll
    for (int t = 0; t < 8; ++t) v += acc[t][d] * acc[t][d];
    s[d] = v;
  }
  // reduce across the 16 lanes of each half (xor masks stay within halves)
#pragma unroll
  for (int m = 1; m <= 8; m <<= 1) {
#pragma unroll
    for (int d = 0; d < 8; ++d) s[d] += __shfl_xor(s[d], m, 32);
  }
  if (r == 0) {
#pragma unroll
    for (int d = 0; d < 8; ++d) {
      const float v = s[d] + bias;
      out[row0 + hi * 8 + d] = 1.0f / (1.0f + __expf(-v));
    }
  }
}

// ---------------------------------------------------------------------------
// Inputs (setup_inputs order): x[B,X], u[B,U], W[K], b[1], V[K,NF]
// ---------------------------------------------------------------------------
extern "C" void kernel_launch(void* const* d_in, const int* in_sizes, int n_in,
                              void* d_out, int out_size, void* d_ws, size_t ws_size,
                              hipStream_t stream) {
  (void)in_sizes; (void)n_in; (void)out_size; (void)ws_size;
  const float* x = (const float*)d_in[0];
  const float* u = (const float*)d_in[1];
  const float* W = (const float*)d_in[2];
  const float* b = (const float*)d_in[3];
  const float* V = (const float*)d_in[4];
  float* out = (float*)d_out;
  __bf16* Bp = (__bf16*)d_ws;       // 1.44 MB of workspace

  fm_pack_B<<<dim3(KT_TOTAL, NT), 32, 0, stream>>>(V, W, Bp);
  fm_main<<<B_ROWS / ROWS_PER_WG, 256, 0, stream>>>(x, u, b, Bp, out);
}
